// LSTM_48567490183924
// MI455X (gfx1250) — matrix operand order repaired and checked by reading, not compiled, not run on hardware
//
#include <hip/hip_runtime.h>
#include <cmath>
#include <cstdint>

typedef __attribute__((ext_vector_type(16))) _Float16 v16h;
typedef __attribute__((ext_vector_type(8)))  float    v8f;
typedef __attribute__((ext_vector_type(4)))  float    v4f;

#define HDIM  1024
#define IDIM  1024
#define BDIM  64
#define TDIM  512
#define KTOT  2048            // I + H concatenated reduction dim
#define GROWS 4096            // 4*H gate rows (i,f,o,g stacked)
#define NKC   (KTOT / 32)     // 64 K-chunks of 32
#define NGT   (GROWS / 16)    // 256 gate-row tiles of 16

static const size_t FRAG_HALVES = (size_t)NGT * NKC * 32 * 16; // 8,388,608 halves (16 MB)
#define GSTRIDE ((size_t)64 * NKC * 32 * 16)   // frag halves between gate blocks (g -> g+1)

// ---------------------------------------------------------------------------
// Prep: shuffle [W|U] (fp32) into WMMA A-fragment order as hi/lo f16 pair.
// A-fragment order used here (16-bit A 16x32, contiguous k; B matches):
//   row = lane % 16,  K = e + 16*(lane >> 4)
// Linear fragment index: (((gtile*NKC + kc)*32 + lane)*16 + e)
// ---------------------------------------------------------------------------
__global__ void lstm_prep_weights(const float* __restrict__ Wi, const float* __restrict__ Wf,
                                  const float* __restrict__ Wo, const float* __restrict__ Wg,
                                  const float* __restrict__ Ui, const float* __restrict__ Uf,
                                  const float* __restrict__ Uo, const float* __restrict__ Ug,
                                  _Float16* __restrict__ whi, _Float16* __restrict__ wlo) {
  size_t idx = (size_t)blockIdx.x * blockDim.x + threadIdx.x;
  if (idx >= FRAG_HALVES) return;
  int e    = (int)(idx & 15);
  int lane = (int)((idx >> 4) & 31);
  int kc   = (int)((idx >> 9) & 63);
  int gt   = (int)(idx >> 15);              // 0..255 gate-row tile
  int row  = gt * 16 + (lane & 15);         // 0..4095
  int k    = kc * 32 + e + ((lane >> 4) << 4);
  int g = row >> 10;                        // which gate (i,f,o,g)
  int j = row & 1023;                       // hidden index
  float w;
  if (k < IDIM) {
    const float* W = (g == 0) ? Wi : (g == 1) ? Wf : (g == 2) ? Wo : Wg;
    w = W[(size_t)j * IDIM + k];
  } else {
    const float* U = (g == 0) ? Ui : (g == 1) ? Uf : (g == 2) ? Uo : Ug;
    w = U[(size_t)j * HDIM + (k - IDIM)];
  }
  _Float16 hi = (_Float16)w;
  whi[idx] = hi;
  wlo[idx] = (_Float16)(w - (float)hi);
}

__global__ void lstm_zero(float* __restrict__ h0, float* __restrict__ h1,
                          float* __restrict__ c) {
  int i = blockIdx.x * blockDim.x + threadIdx.x;
  if (i < BDIM * HDIM) { h0[i] = 0.f; h1[i] = 0.f; c[i] = 0.f; }
}

// ---------------------------------------------------------------------------
// One LSTM timestep.
// Grid: (64 jt, 2 bh) = 128 blocks; block jt owns hidden [jt*16, +16),
// bh owns batches [bh*32, +32). Block: 256 threads = 8 waves;
// wave = (bt2 = wave>>2 in {0,1}, ks = wave&3).
// Each wave converts B (batch cols bh*32+bt2*16..+15) ONCE per K-chunk and
// runs it against all 4 gate A-tiles -> 12 WMMAs per conversion.
// K split 4 ways (ks); slices 0-1 are x_t, 2-3 are h (wave-uniform source).
// Partials reduced over ks in 4 deterministic LDS rounds (no float atomics).
// Doubling blocks over batch is L2-traffic-neutral but doubles busy WGPs,
// halving the latency of each serial step.
// ---------------------------------------------------------------------------
__global__ __launch_bounds__(256)
void lstm_step(const float* __restrict__ x,
               const _Float16* __restrict__ whi, const _Float16* __restrict__ wlo,
               const float* __restrict__ h_in, float* __restrict__ h_out,
               float* __restrict__ c_state,
               const float* __restrict__ bi, const float* __restrict__ bf,
               const float* __restrict__ bo, const float* __restrict__ bg,
               float* __restrict__ out, int t) {
  __shared__ float gl[4][16][34];           // [gate][row-in-tile][batch-in-block] (+pad)

  const int tid  = threadIdx.x;
  const int lane = tid & 31;
  const int wave = tid >> 5;
  const int bt2  = wave >> 2;               // batch tile within block: 0..1
  const int ks   = wave & 3;                // K-slice 0..3 (16 chunks each)
  const int jt   = blockIdx.x;              // hidden tile 0..63
  const int bh   = blockIdx.y;              // batch half 0..1

  // B operand (activations): lane -> column b, K-half within chunk
  const int bcol  = bh * 32 + bt2 * 16 + (lane & 15);
  const int khalf = (lane >> 4) * 16;
  const float* xrow = x    + ((size_t)bcol * TDIM + t) * IDIM;
  const float* hrow = h_in +  (size_t)bcol * HDIM;
  // Slices 0,1 read x at offset k0; slices 2,3 read h at offset k0-1024.
  const float* srcbase = (ks < 2) ? xrow : (hrow - IDIM);

  const int kc0 = ks * 16;
  v8f acc[4] = {};
  for (int kc = kc0; kc < kc0 + 16; ++kc) {
    // ---- B fragment: 16 consecutive f32 -> hi/lo f16 (once per chunk) ----
    const float* src = srcbase + kc * 32 + khalf;
    float va[16];
    *(v4f*)(&va[0])  = *(const v4f*)(src + 0);
    *(v4f*)(&va[4])  = *(const v4f*)(src + 4);
    *(v4f*)(&va[8])  = *(const v4f*)(src + 8);
    *(v4f*)(&va[12]) = *(const v4f*)(src + 12);
    v16h bhi, blo;
#pragma unroll
    for (int e = 0; e < 16; ++e) {
      _Float16 hi = (_Float16)va[e];
      bhi[e] = hi;
      blo[e] = (_Float16)(va[e] - (float)hi);
    }
    // ---- A fragments for all 4 gates; 3 WMMAs each (split-f16 fp32 emu) ----
    const size_t fb = (((size_t)jt * NKC + kc) * 32 + lane) * 16;
#pragma unroll
    for (int g = 0; g < 4; ++g) {
      const size_t off = fb + (size_t)g * GSTRIDE;
      v16h ahi = *(const v16h*)(whi + off);
      v16h alo = *(const v16h*)(wlo + off);
      acc[g] = __builtin_amdgcn_wmma_f32_16x16x32_f16(false, ahi, false, bhi, (short)0, acc[g], false, false);
      acc[g] = __builtin_amdgcn_wmma_f32_16x16x32_f16(false, ahi, false, blo, (short)0, acc[g], false, false);
      acc[g] = __builtin_amdgcn_wmma_f32_16x16x32_f16(false, alo, false, bhi, (short)0, acc[g], false, false);
    }
  }

  // ---- Deterministic K-slice reduction: 4 ordered LDS accumulation rounds.
  // D layout: VGPR r, lanes 0-15: row=r, col=lane; lanes 16-31: row=r+8, col=lane-16
  {
    const int row0 = (lane >> 4) * 8;
    const int col  = bt2 * 16 + (lane & 15);
    for (int r = 0; r < 4; ++r) {
      if (ks == r) {
#pragma unroll
        for (int g = 0; g < 4; ++g)
#pragma unroll
          for (int rr = 0; rr < 8; ++rr) {
            if (r == 0) gl[g][row0 + rr][col]  = acc[g][rr];
            else        gl[g][row0 + rr][col] += acc[g][rr];
          }
      }
      __syncthreads();
    }
  }

  // ---- Pointwise LSTM cell update: 16 hidden indices x 32 batches. ----
  for (int e = tid; e < 16 * 32; e += 256) {
    int rowin = e >> 5;                     // 0..15
    int bloc  = e & 31;                     // batch within block
    int b2    = bh * 32 + bloc;             // global batch
    int j     = jt * 16 + rowin;            // hidden index
    float gi = gl[0][rowin][bloc] + bi[j];
    float gf = gl[1][rowin][bloc] + bf[j];
    float go = gl[2][rowin][bloc] + bo[j];
    float gg = gl[3][rowin][bloc] + bg[j];
    float iv = 1.f / (1.f + __expf(-gi));
    float fv = 1.f / (1.f + __expf(-gf));
    float ov = 1.f / (1.f + __expf(-go));
    float gv = tanhf(gg);
    size_t cidx = (size_t)b2 * HDIM + j;
    float cv = fv * c_state[cidx] + iv * gv;
    float hv = ov * tanhf(cv);
    c_state[cidx] = cv;
    h_out[cidx]   = hv;
    // hs[b][t][j]: streamed once, never re-read -> non-temporal so the
    // L2-resident weight fragments (33 MB) stay hot across all 512 steps.
    __builtin_nontemporal_store(hv, &out[((size_t)b2 * TDIM + t) * HDIM + j]);
  }
}

__global__ void lstm_tail(const float* __restrict__ hfin, const float* __restrict__ cfin,
                          float* __restrict__ out) {
  int i = blockIdx.x * blockDim.x + threadIdx.x;
  if (i < BDIM * HDIM) {
    out[(size_t)BDIM * TDIM * HDIM + i] = hfin[i];
    out[(size_t)BDIM * TDIM * HDIM + (size_t)BDIM * HDIM + i] = cfin[i];
  }
}

extern "C" void kernel_launch(void* const* d_in, const int* in_sizes, int n_in,
                              void* d_out, int out_size, void* d_ws, size_t ws_size,
                              hipStream_t stream) {
  const float* x  = (const float*)d_in[0];
  const float* Wi = (const float*)d_in[1];
  const float* Wf = (const float*)d_in[2];
  const float* Wo = (const float*)d_in[3];
  const float* Wg = (const float*)d_in[4];
  const float* Ui = (const float*)d_in[5];
  const float* Uf = (const float*)d_in[6];
  const float* Uo = (const float*)d_in[7];
  const float* Ug = (const float*)d_in[8];
  const float* bi = (const float*)d_in[9];
  const float* bf = (const float*)d_in[10];
  const float* bo = (const float*)d_in[11];
  const float* bg = (const float*)d_in[12];
  float* out = (float*)d_out;

  // Workspace layout (34.3 MB): h ping-pong, c, then weight fragments (hi, lo).
  float* ws = (float*)d_ws;
  float* h0 = ws;                           // 64K floats
  float* h1 = ws + 65536;
  float* cS = ws + 131072;
  _Float16* whi = (_Float16*)(ws + 196608);
  _Float16* wlo = whi + FRAG_HALVES;

  lstm_zero<<<(BDIM * HDIM + 255) / 256, 256, 0, stream>>>(h0, h1, cS);
  lstm_prep_weights<<<(unsigned)((FRAG_HALVES + 255) / 256), 256, 0, stream>>>(
      Wi, Wf, Wo, Wg, Ui, Uf, Uo, Ug, whi, wlo);

  dim3 grid(64, 2);
  for (int t = 0; t < TDIM; ++t) {
    const float* hin = (t & 1) ? h1 : h0;
    float*       hou = (t & 1) ? h0 : h1;
    lstm_step<<<grid, 256, 0, stream>>>(x, whi, wlo, hin, hou, cS,
                                        bi, bf, bo, bg, out, t);
  }
  // t=511 wrote h0; c final in cS.
  lstm_tail<<<(BDIM * HDIM + 255) / 256, 256, 0, stream>>>(h0, cS, out);
}
